// MixtureOfExperts_48275432407845
// MI455X (gfx1250) — matrix operand order repaired and checked by reading, not compiled, not run on hardware
//
#include <hip/hip_runtime.h>
#include <hip/hip_bf16.h>
#include <math.h>

// Problem constants (from reference)
#define BB 4
#define SS 2048
#define DD 1024
#define FF 4096
#define EE 8
#define TT (BB * SS)            // 8192 tokens
#define MAXTOK TT               // worst case: all tokens to one expert
#define MTILE 32                // tokens per expert workgroup
#define TILES_PER_E (TT / MTILE)   // 256
#define FFC 2048                // FF chunk held in LDS
#define NCHUNK (FF / FFC)       // 2

typedef __attribute__((ext_vector_type(16))) __bf16          v16bf;
typedef __attribute__((ext_vector_type(16))) unsigned short  v16us;
typedef __attribute__((ext_vector_type(8)))  unsigned short  v8us;
typedef __attribute__((ext_vector_type(8)))  float           v8f;

// fp32 -> bf16 bits, round-to-nearest-even
__device__ __forceinline__ unsigned short f2bf(float f) {
    unsigned int u = __builtin_bit_cast(unsigned int, f);
    u += 0x7FFFu + ((u >> 16) & 1u);
    return (unsigned short)(u >> 16);
}

// Exact GeLU: 0.5*x*(1+erf(x/sqrt(2)))
__device__ __forceinline__ float gelu_exact(float x) {
    return 0.5f * x * (1.0f + erff(x * 0.70710678118654752440f));
}

// A-matrix 16x32 bf16 fragment from LDS (row-major, rowStride in elements).
// lane 0-15: row=lane,    elems 0..7 -> K=k0+0..7,  elems 8..15 -> K=k0+16..23
// lane16-31: row=lane-16, elems 0..7 -> K=k0+8..15, elems 8..15 -> K=k0+24..31
__device__ __forceinline__ v16bf load_a_frag_lds(const unsigned short* s, int rowStride,
                                                 int k0, int lane) {
    int row = lane & 15;
    int kb  = (lane >> 4) * 8;
    const unsigned short* base = s + row * rowStride + k0 + kb;
    v8us c0 = *(const v8us*)(base);        // K = k0+kb+0..7
    v8us c1 = *(const v8us*)(base + 16);   // K = k0+kb+16..23
    v16us a = __builtin_shufflevector(c0, c1, 0, 1, 2, 3, 4, 5, 6, 7,
                                              8, 9, 10, 11, 12, 13, 14, 15);
    return __builtin_bit_cast(v16bf, a);
}

// B-matrix 32x16 bf16 fragment from a TRANSPOSED bf16 weight slab [N x K] (K contiguous).
// lane 0-15: col=lane, K=k0+0..7,16..23 ; lane 16-31: col=lane-16, K=k0+8..15,24..31
__device__ __forceinline__ v16bf load_b_frag_T(const unsigned short* Bt, int ldk,
                                               int n0, int k0, int lane) {
    int n  = n0 + (lane & 15);
    int kb = (lane >> 4) * 8;
    const unsigned short* p = Bt + (size_t)n * ldk + k0 + kb;
    v8us c0 = *(const v8us*)(p);       // K = k0+kb+0..7
    v8us c1 = *(const v8us*)(p + 16);  // K = k0+kb+16..23
    v16us b = __builtin_shufflevector(c0, c1, 0, 1, 2, 3, 4, 5, 6, 7,
                                              8, 9, 10, 11, 12, 13, 14, 15);
    return __builtin_bit_cast(v16bf, b);
}

// ---------------------------------------------------------------------------
// Kernel 0: zero output and expert counters
// ---------------------------------------------------------------------------
__global__ __launch_bounds__(256) void moe_zero_kernel(float* out, int n, int* cnt) {
    int i = blockIdx.x * blockDim.x + threadIdx.x;
    if (i < n) out[i] = 0.0f;
    if (i < EE) cnt[i] = 0;
}

// ---------------------------------------------------------------------------
// Kernel 1: elementwise fp32 -> bf16 (for x)
// ---------------------------------------------------------------------------
__global__ __launch_bounds__(256) void conv_bf16_kernel(const float* __restrict__ in,
                                                        unsigned short* __restrict__ out,
                                                        int n) {
    int i = blockIdx.x * blockDim.x + threadIdx.x;
    if (i < n) out[i] = f2bf(in[i]);
}

// ---------------------------------------------------------------------------
// Kernel 2: batched LDS-tiled transpose + fp32->bf16 for expert weights.
// in : E x [R x C] fp32 (row-major) ; out : E x [C x R] bf16 (row-major)
// grid (C/32, R/32, E), block (32, 8)
// ---------------------------------------------------------------------------
__global__ __launch_bounds__(256) void transpose_bf16_kernel(const float* __restrict__ in,
                                                             unsigned short* __restrict__ out,
                                                             int R, int C) {
    __shared__ float t[32][33];
    const size_t mat = (size_t)R * C;
    const float* src = in + (size_t)blockIdx.z * mat;
    unsigned short* dst = out + (size_t)blockIdx.z * mat;
    const int c0 = blockIdx.x * 32, r0 = blockIdx.y * 32;
    const int tx = threadIdx.x, ty = threadIdx.y;
#pragma unroll
    for (int rr = ty; rr < 32; rr += 8)
        t[rr][tx] = src[(size_t)(r0 + rr) * C + (c0 + tx)];
    __syncthreads();
#pragma unroll
    for (int cc = ty; cc < 32; cc += 8)
        dst[(size_t)(c0 + cc) * R + (r0 + tx)] = f2bf(t[tx][cc]);
}

// ---------------------------------------------------------------------------
// Kernel 3: router. One wave32 per token.
// ---------------------------------------------------------------------------
__global__ __launch_bounds__(256) void moe_router_kernel(const float* __restrict__ x,
                                                         const float* __restrict__ Wr,
                                                         int* __restrict__ cnt,
                                                         int* __restrict__ tok_list,
                                                         float* __restrict__ gate_list) {
    int tok  = (blockIdx.x * blockDim.x + threadIdx.x) >> 5;
    int lane = threadIdx.x & 31;
    if (tok >= TT) return;

    const float* xr = x + (size_t)tok * DD;
    float acc[EE];
#pragma unroll
    for (int e = 0; e < EE; ++e) acc[e] = 0.0f;

    for (int k = lane; k < DD; k += 32) {
        float xv = xr[k];
        const float* wrow = Wr + (size_t)k * EE;
#pragma unroll
        for (int e = 0; e < EE; ++e) acc[e] = fmaf(xv, wrow[e], acc[e]);
    }
#pragma unroll
    for (int e = 0; e < EE; ++e) {
#pragma unroll
        for (int off = 16; off > 0; off >>= 1)
            acc[e] += __shfl_xor(acc[e], off, 32);
    }

    if (lane == 0) {
        float m = acc[0];
#pragma unroll
        for (int e = 1; e < EE; ++e) m = fmaxf(m, acc[e]);
        float p[EE];
        float s = 0.0f;
#pragma unroll
        for (int e = 0; e < EE; ++e) { p[e] = __expf(acc[e] - m); s += p[e]; }
        float inv = 1.0f / s;
#pragma unroll
        for (int e = 0; e < EE; ++e) p[e] *= inv;
        int i0 = 0;
#pragma unroll
        for (int e = 1; e < EE; ++e) if (p[e] > p[i0]) i0 = e;
        int i1 = (i0 == 0) ? 1 : 0;
#pragma unroll
        for (int e = 0; e < EE; ++e) if (e != i0 && p[e] > p[i1]) i1 = e;
        float mx = fmaxf(p[i0], p[i1]);
        float e0 = __expf(p[i0] - mx);
        float e1 = __expf(p[i1] - mx);
        float g0 = e0 / (e0 + e1);
        float g1 = e1 / (e0 + e1);

        int pos0 = atomicAdd(&cnt[i0], 1);
        tok_list[i0 * MAXTOK + pos0]  = tok;
        gate_list[i0 * MAXTOK + pos0] = g0;
        int pos1 = atomicAdd(&cnt[i1], 1);
        tok_list[i1 * MAXTOK + pos1]  = tok;
        gate_list[i1 * MAXTOK + pos1] = g1;
    }
}

// ---------------------------------------------------------------------------
// Kernel 4: fused expert FFN over a 32-token tile. 512 threads (16 waves).
//   for each FF chunk (2 x 2048):
//     h = gelu(A @ W1t + b1)           (bf16 in LDS, 32 x 2048)
//     acc2 += h @ W2t                  (persistent per-wave accumulators)
//   out += (acc2 + b2) * gate          (atomic scatter-add)
// A tile gathered into LDS with global_load_async_to_lds_b128 (ASYNCcnt).
// K-loops use unroll(1) + explicit A-frag reuse to avoid VGPR spills.
// ---------------------------------------------------------------------------
__global__ __launch_bounds__(512) void moe_expert_kernel(const unsigned short* __restrict__ xbf,
                                                         const unsigned short* __restrict__ W1t,
                                                         const float* __restrict__ b1,
                                                         const unsigned short* __restrict__ W2t,
                                                         const float* __restrict__ b2,
                                                         const int* __restrict__ cnt,
                                                         const int* __restrict__ tok_list,
                                                         const float* __restrict__ gate_list,
                                                         float* __restrict__ out) {
    __shared__ unsigned short s_A[MTILE * DD];    // 64 KB  : gathered x tile, bf16
    __shared__ unsigned short s_H[MTILE * FFC];   // 128 KB : hidden chunk, bf16
    __shared__ int   s_tok[MTILE];
    __shared__ float s_gate[MTILE];

    const int e     = blockIdx.x >> 8;            // TILES_PER_E == 256
    const int tile  = blockIdx.x & 255;
    const int count = cnt[e];
    if (tile * MTILE >= count) return;            // uniform early-out

    const int tid  = threadIdx.x;
    const int lane = tid & 31;
    const int wv   = tid >> 5;                    // 16 waves

    if (tid < MTILE) {
        int idx = tile * MTILE + tid;
        if (idx < count) {
            s_tok[tid]  = tok_list[e * MAXTOK + idx];
            s_gate[tid] = gate_list[e * MAXTOK + idx];
        } else {
            s_tok[tid]  = -1;
            s_gate[tid] = 0.0f;
        }
    }
    __syncthreads();

    // ---- async gather: 32 bf16 token rows (2 KB each) -> LDS ----
    {
        const unsigned ldsA = (unsigned)(uintptr_t)(void*)s_A;
#pragma unroll
        for (int rr = 0; rr < 2; ++rr) {
            const int row = wv * 2 + rr;
            const int tk  = s_tok[row];
            if (tk >= 0) {
                unsigned long long ga =
                    (unsigned long long)(uintptr_t)(xbf + (size_t)tk * DD) + lane * 16;
                unsigned lo = ldsA + (unsigned)(row * DD * 2 + lane * 16);
#pragma unroll
                for (int it = 0; it < 4; ++it) {   // 4 x (32 lanes x 16B) = 2048 B
                    asm volatile("global_load_async_to_lds_b128 %0, %1, off"
                                 :: "v"(lo + it * 512u), "v"(ga + it * 512ull)
                                 : "memory");
                }
            } else {
                v8us z = {};
#pragma unroll
                for (int it = 0; it < 4; ++it)
                    *(v8us*)(&s_A[row * DD + lane * 8 + it * 256]) = z;
            }
        }
        asm volatile("s_wait_asynccnt 0" ::: "memory");
    }
    __syncthreads();

    const unsigned short* W1te = W1t + (size_t)e * DD * FF;  // [FF x D] bf16
    const unsigned short* W2te = W2t + (size_t)e * FF * DD;  // [D x FF] bf16
    const float* b1e = b1 + (size_t)e * FF;
    const float* b2e = b2 + (size_t)e * DD;

    v8f acc2[8];
#pragma unroll
    for (int i = 0; i < 8; ++i) acc2[i] = (v8f){};

    const int rb1 = wv & 1;                         // GEMM1 row-block is per-wave constant
    const unsigned short* sArb = s_A + rb1 * 16 * DD;

    for (int c = 0; c < NCHUNK; ++c) {
        const int kc = c * FFC;

        // ---- GEMM1 chunk: h[:, kc:kc+FFC] = gelu(A @ W1t + b1) ----
        // wave owns N-tiles nt = (wv + 16*j) >> 1, j = 0..15; 4 concurrent tiles
        for (int j = 0; j < 16; j += 4) {
            v8f acc[4];
            int nt[4];
#pragma unroll
            for (int q = 0; q < 4; ++q) {
                acc[q] = (v8f){};
                nt[q]  = (wv + 16 * (j + q)) >> 1;
            }
#pragma unroll 1
            for (int k0 = 0; k0 < DD; k0 += 32) {
                v16bf a = load_a_frag_lds(sArb, DD, k0, lane);
#pragma unroll
                for (int q = 0; q < 4; ++q) {
                    v16bf b = load_b_frag_T(W1te, DD, kc + nt[q] * 16, k0, lane);
                    acc[q] = __builtin_amdgcn_wmma_f32_16x16x32_bf16(false, a, false, b,
                                                                     (short)0, acc[q],
                                                                     false, false);
                }
            }
            const int rbase = (lane >> 4) * 8;
#pragma unroll
            for (int q = 0; q < 4; ++q) {
                const int colLane = kc + nt[q] * 16 + (lane & 15);
                const float bias  = b1e[colLane];
                const int hcol    = nt[q] * 16 + (lane & 15);
#pragma unroll
                for (int r = 0; r < 8; ++r) {
                    float h = gelu_exact(acc[q][r] + bias);
                    s_H[(rb1 * 16 + rbase + r) * FFC + hcol] = f2bf(h);
                }
            }
        }
        __syncthreads();

        // ---- GEMM2 partial: acc2 += H_chunk @ W2t[:, kc:kc+FFC]^T ----
        // k0 outer; both row-block A-frags loaded once, 8 WMMAs reuse them
#pragma unroll 1
        for (int k0 = 0; k0 < FFC; k0 += 32) {
            v16bf a0 = load_a_frag_lds(s_H, FFC, k0, lane);
            v16bf a1 = load_a_frag_lds(s_H + 16 * FFC, FFC, k0, lane);
#pragma unroll
            for (int i = 0; i < 8; ++i) {
                const int widx = wv + (i << 4);
                const int nt   = widx & 63;
                const int rb   = widx >> 6;
                v16bf b = load_b_frag_T(W2te, FF, nt * 16, kc + k0, lane);
                acc2[i] = __builtin_amdgcn_wmma_f32_16x16x32_bf16(false, rb ? a1 : a0,
                                                                  false, b, (short)0,
                                                                  acc2[i], false, false);
            }
        }
        __syncthreads();   // protect s_H before next chunk overwrites it
    }

    // ---- epilogue: bias, gate, scatter-add ----
#pragma unroll
    for (int i = 0; i < 8; ++i) {
        const int widx  = wv + (i << 4);
        const int nt    = widx & 63;
        const int rb    = widx >> 6;
        const int colN  = nt * 16 + (lane & 15);
        const float bias = b2e[colN];
        const int rbase  = (lane >> 4) * 8;
#pragma unroll
        for (int r = 0; r < 8; ++r) {
            const int row = rb * 16 + rbase + r;
            const int tk  = s_tok[row];
            if (tk >= 0) {
                float v = (acc2[i][r] + bias) * s_gate[row];
                atomicAdd(&out[(size_t)tk * DD + colN], v);
            }
        }
    }
}

// ---------------------------------------------------------------------------
extern "C" void kernel_launch(void* const* d_in, const int* in_sizes, int n_in,
                              void* d_out, int out_size, void* d_ws, size_t ws_size,
                              hipStream_t stream) {
    const float* x   = (const float*)d_in[0];
    const float* Wr  = (const float*)d_in[1];
    const float* W1  = (const float*)d_in[2];
    const float* b1  = (const float*)d_in[3];
    const float* W2  = (const float*)d_in[4];
    const float* b2  = (const float*)d_in[5];
    float* out = (float*)d_out;

    // workspace layout
    char* ws = (char*)d_ws;
    size_t o = 0;
    int* cnt = (int*)(ws + o);             o += 256;
    int* tok_list = (int*)(ws + o);        o += (size_t)EE * MAXTOK * sizeof(int);
    float* gate_list = (float*)(ws + o);   o += (size_t)EE * MAXTOK * sizeof(float);
    o = (o + 255) & ~(size_t)255;
    unsigned short* xbf = (unsigned short*)(ws + o);  o += (size_t)TT * DD * 2;
    unsigned short* W1t = (unsigned short*)(ws + o);  o += (size_t)EE * DD * FF * 2;
    unsigned short* W2t = (unsigned short*)(ws + o);  o += (size_t)EE * FF * DD * 2;

    const int n_out = BB * SS * DD;
    moe_zero_kernel<<<(n_out + 255) / 256, 256, 0, stream>>>(out, n_out, cnt);

    const int n_x = TT * DD;
    conv_bf16_kernel<<<(n_x + 255) / 256, 256, 0, stream>>>(x, xbf, n_x);

    // W1: E x [D x FF] -> W1t: E x [FF x D] bf16
    transpose_bf16_kernel<<<dim3(FF / 32, DD / 32, EE), dim3(32, 8), 0, stream>>>(W1, W1t, DD, FF);
    // W2: E x [FF x D] -> W2t: E x [D x FF] bf16
    transpose_bf16_kernel<<<dim3(DD / 32, FF / 32, EE), dim3(32, 8), 0, stream>>>(W2, W2t, FF, DD);

    moe_router_kernel<<<TT / 8, 256, 0, stream>>>(x, Wr, cnt, tok_list, gate_list);

    moe_expert_kernel<<<EE * TILES_PER_E, 512, 0, stream>>>(xbf, W1t, b1, W2t, b2,
                                                            cnt, tok_list, gate_list, out);
}
